// UVSelfAttention_87737591923351
// MI455X (gfx1250) — compile-verified, matrix operand order checked by reading, not run on hardware
//
#include <hip/hip_runtime.h>
#include <hip/hip_bf16.h>

// ---------------------------------------------------------------------------
// MS-Deformable attention for MI455X (gfx1250, wave32, WMMA).
// B=8, NQ=8192, E=256, heads=8, levels=1, points=4, dh=32, value grid 128x128.
// ---------------------------------------------------------------------------

#define BATCH   8
#define NQUERY  8192
#define EDIM    256
#define HEADS   8
#define POINTS  4
#define DHEAD   32
#define IMG_H   128
#define IMG_W   128
#define NVAL    (IMG_H * IMG_W)   // 16384

typedef _Float16 v16h __attribute__((ext_vector_type(16)));
typedef _Float16 h4   __attribute__((ext_vector_type(4)));
typedef float    v8f  __attribute__((ext_vector_type(8)));

__device__ __forceinline__ h4 pack4(float a, float b, float c, float d) {
  h4 r;
  r[0] = (_Float16)a; r[1] = (_Float16)b;
  r[2] = (_Float16)c; r[3] = (_Float16)d;
  return r;
}

// ---------------------------------------------------------------------------
// Generic GEMM:  C[M,N] = A[M,256] * W[256,N] + bias[N] (+ resid[M,N])
// A, W, bias, resid are f32; compute in f16 WMMA with f32 accumulation.
// Block = 256 threads = 8 waves; block tile 64(M) x 32(N); wave tile 16x16.
// Double-buffered LDS pipeline: chunk i+1 global loads issue before the WMMA
// on chunk i, one barrier per chunk, packed ds_store_b64 staging.
// ---------------------------------------------------------------------------
template <typename OutT, bool RES>
__global__ __launch_bounds__(256) void gemm_wmma_f16(
    const float* __restrict__ A, const float* __restrict__ W,
    const float* __restrict__ bias, const float* __restrict__ resid,
    OutT* __restrict__ C, int N) {
  // rows padded to 40 halfs (80B): 8B-aligned packed stores, bank-skewed
  __shared__ _Float16 sA[2][64][40];   // [buf][m][k]
  __shared__ _Float16 sWt[2][32][40];  // [buf][n][k] (transposed stage)

  const int t    = threadIdx.x;
  const int lane = t & 31;
  const int w    = t >> 5;
  const int m0   = blockIdx.x * 64;
  const int n0   = blockIdx.y * 32;
  const int mw   = (w & 3) * 16;   // wave row offset inside block tile
  const int nw   = (w >> 2) * 16;  // wave col offset inside block tile
  const int lid  = lane & 15;
  const int half = lane >> 4;

  // staging assignments
  const int arow = t >> 3;         // 0..31 (and +32 for second half)
  const int ac4  = (t & 7) << 2;   // k sub-offset 0,4,...,28
  const int wn   = t & 31;         // column within N tile
  const int wk4  = (t >> 5) << 2;  // k sub-offset 0,4,...,28

  const float* Ap0 = A + (size_t)(m0 + arow) * EDIM + ac4;
  const float* Ap1 = A + (size_t)(m0 + arow + 32) * EDIM + ac4;
  const float* Wp  = W + (size_t)wk4 * N + n0 + wn;

  v8f acc = {};

  // ---- prologue: chunk 0 ----
  float4 fa0 = *(const float4*)(Ap0);
  float4 fa1 = *(const float4*)(Ap1);
  float  fw0 = Wp[0];
  float  fw1 = Wp[(size_t)1 * N];
  float  fw2 = Wp[(size_t)2 * N];
  float  fw3 = Wp[(size_t)3 * N];
  *(h4*)&sA[0][arow][ac4]      = pack4(fa0.x, fa0.y, fa0.z, fa0.w);
  *(h4*)&sA[0][arow + 32][ac4] = pack4(fa1.x, fa1.y, fa1.z, fa1.w);
  *(h4*)&sWt[0][wn][wk4]       = pack4(fw0, fw1, fw2, fw3);
  __syncthreads();

  int buf = 0;
#pragma unroll
  for (int i = 0; i < 8; ++i) {
    // ---- issue global loads for chunk i+1 (latency hidden by WMMA) ----
    if (i < 7) {
      const int kn = (i + 1) * 32;
      fa0 = *(const float4*)(Ap0 + kn);
      fa1 = *(const float4*)(Ap1 + kn);
      fw0 = Wp[(size_t)(kn + 0) * N];
      fw1 = Wp[(size_t)(kn + 1) * N];
      fw2 = Wp[(size_t)(kn + 2) * N];
      fw3 = Wp[(size_t)(kn + 3) * N];
      if (i < 6) __builtin_prefetch(Ap0 + (i + 2) * 32, 0, 1);
    }

    // ---- fragments per the 16-bit WMMA VGPR layout, then multiply ----
    // lane (M or N) = lid; per lane K = {half*8 + 0..7} U {16 + half*8 + 0..7}
    {
      const _Float16* ap = &sA[buf][mw + lid][half * 8];
      const _Float16* bp = &sWt[buf][nw + lid][half * 8];
      v16h a, b;
#pragma unroll
      for (int e = 0; e < 8; ++e) {
        a[e]     = ap[e];
        a[e + 8] = ap[16 + e];
        b[e]     = bp[e];
        b[e + 8] = bp[16 + e];
      }
      acc = __builtin_amdgcn_wmma_f32_16x16x32_f16(
          /*neg_a=*/false, a, /*neg_b=*/false, b,
          /*c_mod=*/(short)0, acc, /*reuse_a=*/false, /*reuse_b=*/false);
    }

    // ---- stage chunk i+1 into the other buffer, single barrier ----
    if (i < 7) {
      *(h4*)&sA[buf ^ 1][arow][ac4]      = pack4(fa0.x, fa0.y, fa0.z, fa0.w);
      *(h4*)&sA[buf ^ 1][arow + 32][ac4] = pack4(fa1.x, fa1.y, fa1.z, fa1.w);
      *(h4*)&sWt[buf ^ 1][wn][wk4]       = pack4(fw0, fw1, fw2, fw3);
      __syncthreads();
      buf ^= 1;
    }
  }

  // ---- write D: VGPR r -> row (mw + r) / (mw + 8 + r), col = lane&15 ----
  const int   col   = n0 + nw + lid;
  const int   rbase = m0 + mw + half * 8;
  const float bv    = bias[col];
#pragma unroll
  for (int r = 0; r < 8; ++r) {
    const size_t idx = (size_t)(rbase + r) * N + col;
    float val = acc[r] + bv;
    if (RES) val += resid[idx];
    C[idx] = (OutT)val;
  }
}

// ---------------------------------------------------------------------------
// Softmax over the 4 points + sampling-location precompute.
// One thread per (b*NQ, head). Stores pixel-space coords: s = ref*W + off - 0.5
// ---------------------------------------------------------------------------
__global__ __launch_bounds__(256) void locattn_kernel(
    const float* __restrict__ off, const float* __restrict__ logits,
    const float* __restrict__ ref2d, float* __restrict__ loc,
    float* __restrict__ attn) {
  const int t  = blockIdx.x * 256 + threadIdx.x;  // = bq*HEADS + h
  const int h  = t & (HEADS - 1);
  const int bq = t >> 3;

  const float* lg = logits + (size_t)bq * 32 + h * 4;
  const float l0 = lg[0], l1 = lg[1], l2 = lg[2], l3 = lg[3];
  const float m  = fmaxf(fmaxf(l0, l1), fmaxf(l2, l3));
  const float e0 = __expf(l0 - m), e1 = __expf(l1 - m);
  const float e2 = __expf(l2 - m), e3 = __expf(l3 - m);
  const float inv = 1.0f / (e0 + e1 + e2 + e3);

  const float rx = ref2d[(size_t)bq * 2 + 0] * (float)IMG_W - 0.5f;
  const float ry = ref2d[(size_t)bq * 2 + 1] * (float)IMG_H - 0.5f;

  const float* of = off + (size_t)bq * 64 + h * 8;
  float* lo = loc + (size_t)t * 8;
  float* at = attn + (size_t)t * 4;
  at[0] = e0 * inv; at[1] = e1 * inv; at[2] = e2 * inv; at[3] = e3 * inv;
#pragma unroll
  for (int p = 0; p < POINTS; ++p) {
    lo[p * 2 + 0] = rx + of[p * 2 + 0];  // off/W * W == off
    lo[p * 2 + 1] = ry + of[p * 2 + 1];
  }
}

// ---------------------------------------------------------------------------
// Bilinear gather + attention-weighted reduce.
// One wave32 per (b,q,h): the 32 lanes cover the 32 head channels, so each
// corner gather is a coalesced 64B read from f16 v (resident in 192MB L2).
// ---------------------------------------------------------------------------
__global__ __launch_bounds__(256) void sample_kernel(
    const _Float16* __restrict__ v, const float* __restrict__ loc,
    const float* __restrict__ attn, float* __restrict__ mid) {
  const int gid  = blockIdx.x * 256 + threadIdx.x;
  const int lane = gid & 31;
  const int wave = gid >> 5;        // = bq*HEADS + h
  const int h    = wave & (HEADS - 1);
  const int bq   = wave >> 3;
  const int b    = bq >> 13;        // NQ = 8192

  const _Float16* vb = v + (size_t)b * NVAL * EDIM + h * DHEAD + lane;
  const float* lo = loc + (size_t)wave * 8;
  const float* at = attn + (size_t)wave * 4;

  float acc = 0.0f;
#pragma unroll
  for (int p = 0; p < POINTS; ++p) {
    const float sx  = lo[p * 2 + 0];
    const float sy  = lo[p * 2 + 1];
    const float wgt = at[p];
    const float x0f = floorf(sx), y0f = floorf(sy);
    const int   x0  = (int)x0f,   y0  = (int)y0f;
    const float wx1 = sx - x0f, wx0 = 1.0f - wx1;
    const float wy1 = sy - y0f, wy0 = 1.0f - wy1;
#pragma unroll
    for (int cy = 0; cy < 2; ++cy) {
      const int yy = y0 + cy;
      if (yy < 0 || yy >= IMG_H) continue;
      const float wy = cy ? wy1 : wy0;
#pragma unroll
      for (int cx = 0; cx < 2; ++cx) {
        const int xx = x0 + cx;
        if (xx < 0 || xx >= IMG_W) continue;
        const float wx = cx ? wx1 : wx0;
        acc += wgt * wy * wx * (float)vb[(size_t)(yy * IMG_W + xx) * EDIM];
      }
    }
  }
  mid[(size_t)bq * EDIM + h * DHEAD + lane] = acc;
}

// ---------------------------------------------------------------------------
extern "C" void kernel_launch(void* const* d_in, const int* in_sizes, int n_in,
                              void* d_out, int out_size, void* d_ws,
                              size_t ws_size, hipStream_t stream) {
  (void)in_sizes; (void)n_in; (void)out_size; (void)ws_size;

  const float* query  = (const float*)d_in[0];   // (B, NQ, E)
  const float* value  = (const float*)d_in[1];   // (B, NV, E)
  const float* ref2d  = (const float*)d_in[2];   // (B, NQ, 1, 2)
  /* d_in[3] spatial_shapes: hardcoded 128x128 */
  const float* W_off  = (const float*)d_in[4];   // (E, 64)
  const float* b_off  = (const float*)d_in[5];
  const float* W_attn = (const float*)d_in[6];   // (E, 32)
  const float* b_attn = (const float*)d_in[7];
  const float* W_val  = (const float*)d_in[8];   // (E, E)
  const float* b_val  = (const float*)d_in[9];
  const float* W_out  = (const float*)d_in[10];  // (E, E)
  const float* b_out  = (const float*)d_in[11];
  float* out = (float*)d_out;                    // (B, NQ, E)

  // ---- workspace layout ----
  constexpr size_t NVE  = (size_t)BATCH * NVAL * EDIM;  // 33,554,432 (v, f16)
  constexpr size_t NQ64 = (size_t)BATCH * NQUERY * 64;  // off / loc
  constexpr size_t NQ32 = (size_t)BATCH * NQUERY * 32;  // logits / attn

  _Float16* v   = (_Float16*)d_ws;
  float* fws    = (float*)((char*)d_ws + NVE * sizeof(_Float16));
  float* off    = fws;
  float* logits = off + NQ64;
  float* loc    = logits + NQ32;
  float* attn   = loc + NQ64;
  float* mid    = attn + NQ32;

  const dim3 blk(256);
  const int Mv = BATCH * NVAL;    // 131072
  const int Mq = BATCH * NQUERY;  // 65536

  // 1) v = value @ W_val + b_val     (store f16; fits in L2 for the gathers)
  gemm_wmma_f16<_Float16, false>
      <<<dim3(Mv / 64, EDIM / 32), blk, 0, stream>>>(value, W_val, b_val,
                                                     nullptr, v, EDIM);
  // 2a) off = query @ W_off + b_off
  gemm_wmma_f16<float, false>
      <<<dim3(Mq / 64, 2), blk, 0, stream>>>(query, W_off, b_off, nullptr,
                                             off, 64);
  // 2b) logits = query @ W_attn + b_attn
  gemm_wmma_f16<float, false>
      <<<dim3(Mq / 64, 1), blk, 0, stream>>>(query, W_attn, b_attn, nullptr,
                                             logits, 32);
  // 3) softmax + sampling locations
  locattn_kernel<<<(Mq * HEADS) / 256, blk, 0, stream>>>(off, logits, ref2d,
                                                         loc, attn);
  // 4) bilinear gather-reduce -> mid
  sample_kernel<<<(Mq * HEADS * 32) / 256, blk, 0, stream>>>(v, loc, attn,
                                                             mid);
  // 5) out = mid @ W_out + b_out + query (residual)
  gemm_wmma_f16<float, true>
      <<<dim3(Mq / 64, EDIM / 32), blk, 0, stream>>>(mid, W_out, b_out, query,
                                                     out, EDIM);
}
